// Retention_14998025798184
// MI455X (gfx1250) — matrix operand order repaired and checked by reading, not compiled, not run on hardware
//
#include <hip/hip_runtime.h>

// ---------------------------------------------------------------------------
// Retention (RetNet-style) forward for MI455X (gfx1250, wave32, WMMA).
// B=2, T=2048, E=512, H=8, D=64.  bf16 WMMA (16x16x32) with f32 accumulate.
//  - x and weights pre-converted to bf16 once (no cvt in GEMM loops)
//  - projection k-loop software-pipelined (double-buffered fragments)
//  - K/V tiles staged in LDS via global_load_async_to_lds_b128 (ASYNCcnt),
//    double-buffered, shared by the 8 waves of a block.
// ---------------------------------------------------------------------------

typedef __attribute__((ext_vector_type(16))) __bf16 v16bf;
typedef __attribute__((ext_vector_type(8)))  float  v8f;

constexpr int kB  = 2;
constexpr int kT  = 2048;
constexpr int kE  = 512;
constexpr int kD  = 64;
constexpr int kBT = kB * kT;   // 4096 rows

// fragment viewed as 8 dwords: dword r holds bf16 elements (2r, 2r+1)
union FragBF {
  v16bf v;
  unsigned int u[8];
};

__device__ inline unsigned short bf_store_bits(float f) {
  union { __bf16 b; unsigned short u; } cv;
  cv.b = (__bf16)f;
  return cv.u;
}

// ---------------------------------------------------------------------------
// Elementwise f32 -> bf16 (8 elements/thread; vectorizes to b128 ld/st + pk cvt)
// ---------------------------------------------------------------------------
__global__ __launch_bounds__(256) void cvt_bf16_kernel(
    const float* __restrict__ src, unsigned short* __restrict__ dst, int n)
{
  const int i = (int)(blockIdx.x * blockDim.x + threadIdx.x) * 8;
  if (i >= n) return;
  union { unsigned short s[8]; uint4 v; } pk;
#pragma unroll
  for (int e = 0; e < 8; ++e) pk.s[e] = bf_store_bits(src[i + e]);
  *(uint4*)(dst + i) = pk.v;
}

// ---------------------------------------------------------------------------
// Projection GEMM on bf16 operands: out[r,c] = sum_k x[r,k]*w[c,k] + bias[c]
// One wave per 16x64 output tile; k-loop double-buffered: loads for chunk
// k+1 are issued before the 4 wmma of chunk k, so L2 latency overlaps the
// matrix pipe instead of draining loadcnt to 0 each step.
// mode 0: bf16 row-major [BT,E].  mode 1: silu f32.  mode 2: bf16 transposed [E,BT].
// ---------------------------------------------------------------------------
__global__ __launch_bounds__(256) void proj_kernel(
    const unsigned short* __restrict__ xb, const unsigned short* __restrict__ wb,
    const float* __restrict__ bias,
    unsigned short* __restrict__ out_bf, float* __restrict__ out_f32,
    int mode)
{
  const int wid  = (int)((blockIdx.x * blockDim.x + threadIdx.x) >> 5);
  const int lane = (int)(threadIdx.x & 31);
  const int rowTile = wid >> 3;   // 4096/16 = 256 row tiles
  const int colGrp  = wid & 7;    // 512/64  = 8 column groups
  const int row0 = rowTile * 16;
  const int col0 = colGrp * 64;
  const int m = lane & 15;        // A-row / B-col index for this lane
  const int g = lane >> 4;        // lane group

  FragBF a[2], bfr[2][4];

  auto loadChunk = [&](int sel, int k0) {
#pragma unroll
    for (int r = 0; r < 8; ++r) {
      const int ka = 2 * r + 16 * g;                     // A 16x32 layout
      a[sel].u[r] = *(const unsigned int*)&xb[(row0 + m) * kE + k0 + ka];
    }
#pragma unroll
    for (int t = 0; t < 4; ++t) {
#pragma unroll
      for (int r = 0; r < 8; ++r) {
        const int kk = 2 * r + 16 * g;                   // B 32x16 layout
        bfr[sel][t].u[r] =
            *(const unsigned int*)&wb[(col0 + t * 16 + m) * kE + k0 + kk];
      }
    }
  };

  v8f c[4] = {};
  loadChunk(0, 0);
#pragma unroll 2
  for (int k0 = 0; k0 < kE; k0 += 32) {
    const int cur = (k0 >> 5) & 1;
    if (k0 + 32 < kE) loadChunk(cur ^ 1, k0 + 32);   // prefetch next chunk
#pragma unroll
    for (int t = 0; t < 4; ++t) {
      c[t] = __builtin_amdgcn_wmma_f32_16x16x32_bf16(
          false, a[cur].v, false, bfr[cur][t].v, (short)0, c[t], false, false);
    }
  }

#pragma unroll
  for (int t = 0; t < 4; ++t) {
    const int cc = col0 + t * 16 + m;       // N = lane % 16
    const float bv = bias[cc];
#pragma unroll
    for (int e = 0; e < 8; ++e) {
      const int rr = row0 + e + 8 * g;      // M = e + 8*(lane>=16)
      const float v = c[t][e] + bv;
      if (mode == 0) {
        out_bf[rr * kE + cc] = bf_store_bits(v);
      } else if (mode == 1) {
        out_f32[rr * kE + cc] = v * __builtin_amdgcn_rcpf(1.f + __expf(-v));
      } else {
        out_bf[cc * kBT + rr] = bf_store_bits(v);   // transposed (for V)
      }
    }
  }
}

// ---------------------------------------------------------------------------
// Retention attention + fused gate.
// Block = 8 waves sharing (b,h) and 256 consecutive rows (wave w owns rows
// [i0Blk + 32w, +32)).  Per 32-wide j-block:
//   - K tile 32x64 and V tile 64x32 (bf16) async-copied to LDS, double-buffered
//   - each wave (when jb >= its i0):
//       S[32x32] = Q[32x64] * K^T                       (8 wmma)
//       S *= gamma^(j-i)/8, ->bf16 -> LDS score bounce
//       O[32x64] += S * V                               (8 wmma)
// Epilogue fuses the silu-gate multiply.
// ---------------------------------------------------------------------------
__global__ __launch_bounds__(256) void retention_attn(
    const unsigned short* __restrict__ qb, const unsigned short* __restrict__ kb,
    const unsigned short* __restrict__ vbT, const float* __restrict__ gate,
    float* __restrict__ out)
{
  __shared__ unsigned short sS[8][32 * 32];   // per-wave score bounce (16 KB)
  __shared__ unsigned short sK[2][32 * 64];   // K tiles, double-buffered (8 KB)
  __shared__ unsigned short sV[2][64 * 32];   // V tiles, double-buffered (8 KB)

  const int tid      = (int)threadIdx.x;
  const int widInBlk = tid >> 5;
  const int lane     = tid & 31;

  const int rowBlk = (int)blockIdx.x & 7;         // 8 row-blocks of 256
  const int h      = ((int)blockIdx.x >> 3) & 7;
  const int b      = (int)blockIdx.x >> 6;
  const int iBlk0  = rowBlk * 256;
  const int i0     = iBlk0 + widInBlk * 32;       // this wave's rows
  const int m = lane & 15, g = lane >> 4;
  const int rowBase = b * kT;
  const int colH    = h * kD;

  // per-thread async-copy duty (16 B each, 256 threads cover a 4 KB tile)
  const int krow = tid >> 3, kcol = (tid & 7) * 8;   // K: 32 rows x 64 halves
  const int vrow = tid >> 2, vcol = (tid & 3) * 8;   // V: 64 rows x 32 halves

  auto issueTile = [&](int bufSel, int jb) {
    const unsigned kd = (unsigned)(size_t)&sK[bufSel][krow * 64 + kcol];
    const void*    ks = (const void*)&kb[(rowBase + jb + krow) * kE + colH + kcol];
    asm volatile("global_load_async_to_lds_b128 %0, %1, off"
                 :: "v"(kd), "v"(ks) : "memory");
    const unsigned vd = (unsigned)(size_t)&sV[bufSel][vrow * 32 + vcol];
    const void*    vs = (const void*)&vbT[(colH + vrow) * kBT + rowBase + jb + vcol];
    asm volatile("global_load_async_to_lds_b128 %0, %1, off"
                 :: "v"(vd), "v"(vs) : "memory");
  };

  // Q A-fragments: [sub 0/1 = rows 0-15/16-31][k-chunk 0/1 = d 0-31/32-63]
  FragBF aq[2][2];
#pragma unroll
  for (int sub = 0; sub < 2; ++sub) {
#pragma unroll
    for (int r = 0; r < 8; ++r) {
      const int ka = 2 * r + 16 * g;
      const unsigned short* p =
          &qb[(rowBase + i0 + sub * 16 + m) * kE + colH + ka];
      aq[sub][0].u[r] = *(const unsigned int*)(p);
      aq[sub][1].u[r] = *(const unsigned int*)(p + 32);
    }
  }

  const float gamma = 1.f - exp2f(-5.f - (float)h);
  const float lg2   = log2f(gamma);

  v8f o[2][4] = {};                  // 32x64 output accumulator
  unsigned short* myS = sS[widInBlk];

  issueTile(0, iBlk0);               // prologue copy
  int buf = 0;
  for (int jb = iBlk0; jb < kT; jb += 32) {
    asm volatile("s_wait_asynccnt 0x0" ::: "memory");  // my copies for `buf` done
    __syncthreads();                                   // whole tile visible

    if (jb + 32 < kT) issueTile(buf ^ 1, jb + 32);     // overlap copy w/ compute

    if (jb >= i0) {                  // wave-uniform guard (EXEC stays full)
      // ---- K fragments from LDS (shared by both row subtiles) ----
      FragBF bkf[2][2];              // [nh = j 0-15/16-31][k-chunk]
#pragma unroll
      for (int nh = 0; nh < 2; ++nh) {
#pragma unroll
        for (int r = 0; r < 8; ++r) {
          const int kk = 2 * r + 16 * g;
          const unsigned short* p = &sK[buf][(nh * 16 + m) * 64 + kk];
          bkf[nh][0].u[r] = *(const unsigned int*)(p);
          bkf[nh][1].u[r] = *(const unsigned int*)(p + 32);
        }
      }

      // ---- scores, decay, bf16, stash to score-bounce LDS ----
#pragma unroll
      for (int sub = 0; sub < 2; ++sub) {
#pragma unroll
        for (int nh = 0; nh < 2; ++nh) {
          v8f s = {};
          s = __builtin_amdgcn_wmma_f32_16x16x32_bf16(false, aq[sub][0].v, false,
                                                      bkf[nh][0].v, (short)0, s,
                                                      false, false);
          s = __builtin_amdgcn_wmma_f32_16x16x32_bf16(false, aq[sub][1].v, false,
                                                      bkf[nh][1].v, (short)0, s,
                                                      false, false);
#pragma unroll
          for (int e = 0; e < 8; ++e) {
            const int mi = sub * 16 + e + 8 * g;           // row within 32
            const int dj = (jb + nh * 16 + m) - (i0 + mi); // j - i
            const float f = (dj >= 0) ? exp2f(lg2 * (float)dj) * 0.125f : 0.f;
            myS[mi * 32 + (nh * 16 + m)] = bf_store_bits(s[e] * f);
          }
        }
      }
      asm volatile("s_wait_dscnt 0x0" ::: "memory");   // wave-private bounce

      // ---- re-fragment S (two bf16 16x32 A operands) ----
      FragBF as[2];
#pragma unroll
      for (int sub = 0; sub < 2; ++sub) {
#pragma unroll
        for (int r = 0; r < 8; ++r) {
          const int ka = 2 * r + 16 * g;
          as[sub].u[r] = *(const unsigned int*)&myS[(sub * 16 + m) * 32 + ka];
        }
      }

      // ---- O[32x64] += S * V ; V fragments from LDS ----
#pragma unroll
      for (int t = 0; t < 4; ++t) {
        FragBF bv;
#pragma unroll
        for (int r = 0; r < 8; ++r) {
          const int kk = 2 * r + 16 * g;
          bv.u[r] = *(const unsigned int*)&sV[buf][(t * 16 + m) * 32 + kk];
        }
#pragma unroll
        for (int sub = 0; sub < 2; ++sub) {
          o[sub][t] = __builtin_amdgcn_wmma_f32_16x16x32_bf16(
              false, as[sub].v, false, bv.v, (short)0, o[sub][t], false, false);
        }
      }
    }
    buf ^= 1;
  }

  // ---- epilogue: fused gate multiply ----
#pragma unroll
  for (int sub = 0; sub < 2; ++sub) {
#pragma unroll
    for (int t = 0; t < 4; ++t) {
#pragma unroll
      for (int e = 0; e < 8; ++e) {
        const int rr = rowBase + i0 + sub * 16 + e + 8 * g;
        const int cc = colH + t * 16 + m;
        out[rr * kE + cc] = gate[rr * kE + cc] * o[sub][t][e];
      }
    }
  }
}

// ---------------------------------------------------------------------------
extern "C" void kernel_launch(void* const* d_in, const int* in_sizes, int n_in,
                              void* d_out, int out_size, void* d_ws, size_t ws_size,
                              hipStream_t stream) {
  const float* x  = (const float*)d_in[0];
  const float* wq = (const float*)d_in[1];
  const float* bq = (const float*)d_in[2];
  const float* wk = (const float*)d_in[3];
  const float* bk = (const float*)d_in[4];
  const float* wv = (const float*)d_in[5];
  const float* bv = (const float*)d_in[6];
  const float* wg = (const float*)d_in[7];
  const float* bg = (const float*)d_in[8];
  float* out = (float*)d_out;

  // workspace layout:
  //   qb  [BT*E] bf16        4 MB
  //   kbw [BT*E] bf16        4 MB
  //   vbT [E*BT] bf16 (V^T)  4 MB
  //   gate[BT*E] f32         8 MB
  //   xb  [BT*E] bf16        4 MB   (x pre-converted)
  //   wb  [E*E]  bf16        0.5 MB (weight staging, reused serially)
  unsigned short* qb  = (unsigned short*)d_ws;
  unsigned short* kbw = qb  + (size_t)kBT * kE;
  unsigned short* vbT = kbw + (size_t)kBT * kE;
  float*          gate = (float*)(vbT + (size_t)kBT * kE);
  unsigned short* xb  = (unsigned short*)(gate + (size_t)kBT * kE);
  unsigned short* wb  = xb + (size_t)kBT * kE;

  const int nX = kBT * kE;   // 2M elements
  const int nW = kE * kE;    // 256K elements
  dim3 cblkX((nX / 8 + 255) / 256), cblkW((nW / 8 + 255) / 256);

  cvt_bf16_kernel<<<cblkX, 256, 0, stream>>>(x, xb, nX);

  // 256 row tiles * 8 col groups = 2048 waves -> 256 blocks of 256
  dim3 pblk(256), pgrd(256);
  cvt_bf16_kernel<<<cblkW, 256, 0, stream>>>(wq, wb, nW);
  proj_kernel<<<pgrd, pblk, 0, stream>>>(xb, wb, bq, qb,  nullptr, 0);
  cvt_bf16_kernel<<<cblkW, 256, 0, stream>>>(wk, wb, nW);
  proj_kernel<<<pgrd, pblk, 0, stream>>>(xb, wb, bk, kbw, nullptr, 0);
  cvt_bf16_kernel<<<cblkW, 256, 0, stream>>>(wv, wb, nW);
  proj_kernel<<<pgrd, pblk, 0, stream>>>(xb, wb, bv, vbT, nullptr, 2);
  cvt_bf16_kernel<<<cblkW, 256, 0, stream>>>(wg, wb, nW);
  proj_kernel<<<pgrd, pblk, 0, stream>>>(xb, wb, bg, nullptr, gate, 1);

  // attention: 2 b * 8 h * 8 row-blocks = 128 blocks of 256 (8 waves)
  retention_attn<<<128, 256, 0, stream>>>(qb, kbw, vbT, gate, out);
}
